// GCNWithNFFNN_4733053960352
// MI455X (gfx1250) — matrix operand order repaired and checked by reading, not compiled, not run on hardware
//
#include <hip/hip_runtime.h>
#include <hip/hip_bf16.h>

#define DD 128
#define NG 64

typedef __bf16 bf16_t;
typedef __attribute__((ext_vector_type(16))) __bf16 v16bf;
typedef __attribute__((ext_vector_type(8)))  float  v8f;
typedef __attribute__((ext_vector_type(4)))  float  v4f;
typedef __attribute__((ext_vector_type(4)))  unsigned int v4u;

// ---------------- elementwise / setup kernels ----------------

__global__ void zero_f32(float* __restrict__ p, int n) {
  int i = blockIdx.x * blockDim.x + threadIdx.x;
  if (i < n) p[i] = 0.0f;
}

__global__ void degree_kernel(const int* __restrict__ src, const int* __restrict__ dst,
                              float* __restrict__ deg_out, float* __restrict__ deg_in, int E) {
  int e = blockIdx.x * blockDim.x + threadIdx.x;
  if (e < E) {
    atomicAdd(&deg_out[src[e]], 1.0f);
    atomicAdd(&deg_in[dst[e]], 1.0f);
  }
}

__global__ void norm_kernel(float* __restrict__ p, int n) {
  int i = blockIdx.x * blockDim.x + threadIdx.x;
  if (i < n) p[i] = rsqrtf(fmaxf(p[i], 1.0f));
}

// out[i] = in[i] * norm[row(i)]  (in==out allowed: pure elementwise)
__global__ void scale_rows(const float* __restrict__ in, const float* __restrict__ norm,
                           float* __restrict__ out, int total) {
  int i = blockIdx.x * blockDim.x + threadIdx.x;
  if (i < total) out[i] = in[i] * norm[i >> 7];
}

// ---------------- SpMM: agg[dst[e]] += hs[src[e]]  (one wave32 per edge, float4/lane)
// src/dst indices forced into SGPRs via readfirstlane -> saddr-form loads/atomics.
__global__ void spmm_kernel(const float* __restrict__ hs, const int* __restrict__ src,
                            const int* __restrict__ dst, float* __restrict__ agg, int E) {
  int wave = (blockIdx.x * blockDim.x + threadIdx.x) >> 5;
  int lane = threadIdx.x & 31;
  if (wave >= E) return;
  int s = __builtin_amdgcn_readfirstlane(src[wave]);
  int d = __builtin_amdgcn_readfirstlane(dst[wave]);
  const v4f v = *(const v4f*)(hs + (size_t)s * DD + lane * 4);
  float* o = agg + (size_t)d * DD + lane * 4;
  atomicAdd(o + 0, v.x);
  atomicAdd(o + 1, v.y);
  atomicAdd(o + 2, v.z);
  atomicAdd(o + 3, v.w);
}

// ---------------- weight transpose + bf16 convert: Wt[n*128+k] = bf16(W[k*128+n])
__global__ void convert_w_kernel(const float* __restrict__ W, bf16_t* __restrict__ Wt) {
  int i = blockIdx.x * blockDim.x + threadIdx.x;
  if (i < DD * DD) {
    int k = i >> 7, n = i & 127;
    Wt[n * DD + k] = (bf16_t)W[k * DD + n];
  }
}

// ---------------- WMMA GEMM: out = act( (rownorm? diag(rownorm): I) * A @ W + b )
// A: [M,128] f32, Wt: [128,128] bf16 N-major (row n holds K=0..127), out: [M,128] f32.
// REQUIRES: M % 16 == 0 and grid.x == M/16 (no bounds checks -> straight-line stores).
// 256 threads = 8 wave32; block owns 16 rows; wave w owns columns w*16..w*16+15.
__global__ __launch_bounds__(256) void wmma_gemm_kernel(
    const float* __restrict__ A, const float* __restrict__ rownorm,
    const bf16_t* __restrict__ Wt, const float* __restrict__ bias,
    float* __restrict__ out, int relu) {
  __shared__ bf16_t lds_a[16 * DD];  // 4 KB: 16-row bf16 slab

  const int tid = threadIdx.x;
  const int m0  = blockIdx.x * 16;

  // Stage A tile: scale rows by norm (or 1), convert f32 -> bf16 into LDS.
#pragma unroll
  for (int j = 0; j < 8; ++j) {
    int idx = tid + j * 256;          // 0..2047
    int r   = idx >> 7;
    int ccl = idx & 127;
    int row = m0 + r;
    float v = A[(size_t)row * DD + ccl];
    if (rownorm != nullptr) v *= rownorm[row];
    lds_a[idx] = (bf16_t)v;
  }
  __syncthreads();

  const int wave  = tid >> 5;   // N tile index (0..7)
  const int lane  = tid & 31;
  const int khalf = lane >> 4;  // 0: lanes 0-15, 1: lanes 16-31
  const int l15   = lane & 15;

  // B source: Wt row n (= 128 contiguous bf16 K values)
  const bf16_t* wrow = Wt + (size_t)(wave * 16 + l15) * DD;

  v8f acc = {0.f, 0.f, 0.f, 0.f, 0.f, 0.f, 0.f, 0.f};

#pragma unroll
  for (int kk = 0; kk < 4; ++kk) {   // K = kk*32 .. kk*32+31
    union { v16bf v; v4u q[2]; } a;
    union { v16bf v; v4u q[2]; } b;
    // A 16x32 bf16 layout: elems 0..7  -> K = kk*32 + khalf*8 + {0..7}
    //                      elems 8..15 -> K = kk*32 + 16 + khalf*8 + {0..7}
    const bf16_t* ap = lds_a + l15 * DD + kk * 32 + khalf * 8;
    a.q[0] = *(const v4u*)(ap);
    a.q[1] = *(const v4u*)(ap + 16);
    // B 32x16 bf16 layout: same K order as A, N = l15
    const v4u* bp = (const v4u*)(wrow + kk * 32 + khalf * 8);
    b.q[0] = bp[0];
    b.q[1] = bp[2];
    acc = __builtin_amdgcn_wmma_f32_16x16x32_bf16(
        /*neg_a=*/false, a.v, /*neg_b=*/false, b.v,
        /*c_mod=*/(short)0, acc, /*reuse_a=*/false, /*reuse_b=*/false);
  }

  // C/D layout: VGPR r -> M = r + 8*khalf, N = l15 (within tile).
  // Unconditional stores: each writes 2x 64B contiguous segments.
  const int col = wave * 16 + l15;
  const float bv = bias[col];
  float* orow = out + (size_t)(m0 + khalf * 8) * DD + col;
#pragma unroll
  for (int r = 0; r < 8; ++r) {
    float v = acc[r] + bv;
    if (relu) v = fmaxf(v, 0.0f);
    orow[(size_t)r * DD] = v;
  }
}

// ---------------- per-graph mean pooling ----------------

__global__ void pool_kernel(const float* __restrict__ h, const int* __restrict__ gid,
                            float* __restrict__ sums, float* __restrict__ cnt, int N) {
  int i    = blockIdx.x * blockDim.x + threadIdx.x;  // one wave32 per node
  int node = i >> 5;
  int l4   = (i & 31) * 4;
  if (node >= N) return;
  int g = __builtin_amdgcn_readfirstlane(gid[node]);
  const v4f v = *(const v4f*)(h + (size_t)node * DD + l4);
  float* o = sums + g * DD + l4;
  atomicAdd(o + 0, v.x);
  atomicAdd(o + 1, v.y);
  atomicAdd(o + 2, v.z);
  atomicAdd(o + 3, v.w);
  if ((i & 31) == 0) atomicAdd(&cnt[g], 1.0f);
}

__global__ void mean_kernel(const float* __restrict__ sums, const float* __restrict__ cnt,
                            float* __restrict__ out) {
  int i = blockIdx.x * blockDim.x + threadIdx.x;
  if (i < NG * DD) out[i] = sums[i] / fmaxf(cnt[i >> 7], 1.0f);
}

// ---------------- final projection + sigmoid ----------------

__global__ void out_kernel(const float* __restrict__ h, const float* __restrict__ w,
                           const float* __restrict__ b, float* __restrict__ out) {
  int g = threadIdx.x;
  if (g >= NG) return;
  float acc = b[0];
#pragma unroll 4
  for (int k = 0; k < DD; ++k) acc += h[g * DD + k] * w[k];
  out[g] = 1.0f / (1.0f + __expf(-acc));
}

// ---------------- driver ----------------

static inline int cdiv_i(int a, int b) { return (a + b - 1) / b; }

extern "C" void kernel_launch(void* const* d_in, const int* in_sizes, int n_in,
                              void* d_out, int out_size, void* d_ws, size_t ws_size,
                              hipStream_t stream) {
  const float* x      = (const float*)d_in[0];
  const int*   src    = (const int*)d_in[1];
  const int*   dst    = (const int*)d_in[2];
  const int*   gid    = (const int*)d_in[3];
  const float* conv_W = (const float*)d_in[4];
  const float* conv_b = (const float*)d_in[5];
  const float* ffnn_W = (const float*)d_in[6];
  const float* ffnn_b = (const float*)d_in[7];
  const float* out_W  = (const float*)d_in[8];
  const float* out_b  = (const float*)d_in[9];

  const int N = in_sizes[0] / DD;             // 50000 (multiple of 16)
  const int E = in_sizes[1];                  // 600000
  const int NCONV = in_sizes[4] / (DD * DD);  // 4
  const int NFFNN = in_sizes[6] / (DD * DD);  // 3

  // Workspace carving, 256B aligned.
  char* ws = (char*)d_ws;
  size_t off = 0;
  auto carve = [&](size_t bytes) -> void* {
    void* p = ws + off;
    off += (bytes + 255) & ~(size_t)255;
    return p;
  };
  float*  norm_src = (float*)carve((size_t)N * 4);
  float*  norm_dst = (float*)carve((size_t)N * 4);
  float*  S        = (float*)carve((size_t)N * DD * 4);  // scaled feats / layer output
  float*  G        = (float*)carve((size_t)N * DD * 4);  // SpMM accumulator
  bf16_t* Wt       = (bf16_t*)carve((size_t)DD * DD * 2);
  float*  sums     = (float*)carve((size_t)NG * DD * 4);
  float*  cnt      = (float*)carve((size_t)NG * 4);
  float*  fb0      = (float*)carve((size_t)NG * DD * 4);
  float*  fb1      = (float*)carve((size_t)NG * DD * 4);

  // Degrees -> norms
  zero_f32<<<cdiv_i(N, 256), 256, 0, stream>>>(norm_src, N);
  zero_f32<<<cdiv_i(N, 256), 256, 0, stream>>>(norm_dst, N);
  degree_kernel<<<cdiv_i(E, 256), 256, 0, stream>>>(src, dst, norm_src, norm_dst, E);
  norm_kernel<<<cdiv_i(N, 256), 256, 0, stream>>>(norm_src, N);
  norm_kernel<<<cdiv_i(N, 256), 256, 0, stream>>>(norm_dst, N);

  // Graph convolutions
  const float* h = x;
  for (int l = 0; l < NCONV; ++l) {
    scale_rows<<<cdiv_i(N * DD, 256), 256, 0, stream>>>(h, norm_src, S, N * DD);
    zero_f32<<<cdiv_i(N * DD, 256), 256, 0, stream>>>(G, N * DD);
    convert_w_kernel<<<cdiv_i(DD * DD, 256), 256, 0, stream>>>(conv_W + (size_t)l * DD * DD, Wt);
    spmm_kernel<<<cdiv_i(E * 32, 256), 256, 0, stream>>>(S, src, dst, G, E);
    wmma_gemm_kernel<<<N / 16, 256, 0, stream>>>(
        G, norm_dst, Wt, conv_b + (size_t)l * DD, S, (l < NCONV - 1) ? 1 : 0);
    h = S;
  }

  // Mean pooling
  zero_f32<<<cdiv_i(NG * DD, 256), 256, 0, stream>>>(sums, NG * DD);
  zero_f32<<<1, 64, 0, stream>>>(cnt, NG);
  pool_kernel<<<cdiv_i(N * 32, 256), 256, 0, stream>>>(S, gid, sums, cnt, N);
  mean_kernel<<<cdiv_i(NG * DD, 256), 256, 0, stream>>>(sums, cnt, fb0);

  // FFNN stack (WMMA, M = 64)
  float* fin = fb0;
  float* fout = fb1;
  for (int f = 0; f < NFFNN; ++f) {
    convert_w_kernel<<<cdiv_i(DD * DD, 256), 256, 0, stream>>>(ffnn_W + (size_t)f * DD * DD, Wt);
    wmma_gemm_kernel<<<NG / 16, 256, 0, stream>>>(
        fin, nullptr, Wt, ffnn_b + (size_t)f * DD, fout, 1);
    float* t = fin; fin = fout; fout = t;
  }

  // Output projection + sigmoid -> d_out [64]
  out_kernel<<<1, 64, 0, stream>>>(fin, out_W, out_b, (float*)d_out);
}
